// MoE_59442347377056
// MI455X (gfx1250) — matrix operand order repaired and checked by reading, not compiled, not run on hardware
//
#include <hip/hip_runtime.h>

typedef __attribute__((ext_vector_type(16))) __bf16       v16bf;
typedef __attribute__((ext_vector_type(4)))  __bf16       v4bf;
typedef __attribute__((ext_vector_type(8)))  float        v8f;
typedef __attribute__((ext_vector_type(4)))  float        v4f;
typedef __attribute__((ext_vector_type(4)))  unsigned int v4u;
typedef __attribute__((ext_vector_type(2)))  unsigned int v2u;

#define IN_DIM  4096
#define H_DIM   4096
#define OUT_DIM 4096
#define R_DIM   512
#define E_DIM   4
#define B_DIM   64
#define TOT_SZ  ((long)R_DIM*IN_DIM + (long)H_DIM*R_DIM + (long)R_DIM*H_DIM + (long)OUT_DIM*R_DIM)

// Vector fp32->bf16: lowers to packed v_cvt_pk_bf16_f32 (2 instrs for 4 elems,
// no sub-register repacking movs).
__device__ __forceinline__ v2u cvt4(v4f f) {
    v4bf h = __builtin_convertvector(f, v4bf);
    return __builtin_bit_cast(v2u, h);
}

// C_part(64 x N block) = A(64 x K-slice) * (W + delta)^T
//   expert = blockIdx.y, N-block of 128 = blockIdx.x, K-split = blockIdx.z
// Software-pipelined: global loads for step k+1 overlap ds_load+WMMA of step k.
// delta is streamed non-temporal (read exactly once); W/A stay regular (L2
// reuse across experts / N-blocks; L2 = 192 MB easily holds W slices).
template<int RELU>
__global__ __launch_bounds__(256) void lora_gemm(
    const float* __restrict__ A, long strideA,
    const float* __restrict__ W,
    const float* __restrict__ delta, long dOff,
    float* __restrict__ C, long partStride,
    int K, int N, int kLen)
{
    // bf16 tiles staged as dwords; row stride 20 dwords (80 B, 16B-aligned rows)
    __shared__ __align__(16) unsigned ldsA[64 * 20];    // 64 x 32 bf16
    __shared__ __align__(16) unsigned ldsB[128 * 20];   // 128 x 32 bf16

    const int tid  = threadIdx.x;
    const int lane = tid & 31;
    const int wave = tid >> 5;
    const int e      = blockIdx.y;
    const int nBase  = blockIdx.x * 128;
    const int kStart = blockIdx.z * kLen;
    const int kEnd   = kStart + kLen;

    const float* Ae = A + (long)e * strideA;
    const float* De = delta + (long)e * TOT_SZ + dOff;

    const int mt = wave & 3;          // m-tile (rows mt*16 .. +15)
    const int ng = (wave >> 2) * 4;   // first of 4 n-tiles (0 or 4)

    v8f acc[4] = {};

    // Loader mapping (256 threads)
    const int ar = tid >> 2;          // A row 0..63
    const int ac = (tid & 3) * 8;     // A col 0,8,16,24
    const int br = tid >> 1;          // weight row 0..127
    const int bc = (tid & 1) * 16;    // weight col 0,16

    // In-flight stage registers
    v4f a0, a1, w0, w1, w2, w3, d0, d1, d2, d3;

    auto load_stage = [&](int kk) {
        const float* asrc = Ae + (long)ar * K + kk + ac;
        a0 = *(const v4f*)(asrc);
        a1 = *(const v4f*)(asrc + 4);
        long off = (long)(nBase + br) * K + kk + bc;
        const float* w = W  + off;
        const float* d = De + off;
        w0 = *(const v4f*)(w);      w1 = *(const v4f*)(w + 4);
        w2 = *(const v4f*)(w + 8);  w3 = *(const v4f*)(w + 12);
        d0 = __builtin_nontemporal_load((const v4f*)(d));
        d1 = __builtin_nontemporal_load((const v4f*)(d + 4));
        d2 = __builtin_nontemporal_load((const v4f*)(d + 8));
        d3 = __builtin_nontemporal_load((const v4f*)(d + 12));
    };

    auto commit_stage = [&]() {
        v2u pa0 = cvt4(a0), pa1 = cvt4(a1);
        v4u pa = { pa0[0], pa0[1], pa1[0], pa1[1] };
        *(v4u*)(ldsA + ar * 20 + (ac >> 1)) = pa;

        v2u q0 = cvt4(w0 + d0), q1 = cvt4(w1 + d1);
        v2u q2 = cvt4(w2 + d2), q3 = cvt4(w3 + d3);
        v4u p0 = { q0[0], q0[1], q1[0], q1[1] };
        v4u p1 = { q2[0], q2[1], q3[0], q3[1] };
        unsigned* db = ldsB + br * 20 + (bc >> 1);
        *(v4u*)(db)     = p0;
        *(v4u*)(db + 4) = p1;
    };

    load_stage(kStart);

    for (int kk = kStart; kk < kEnd; kk += 32) {
        __syncthreads();          // previous iteration's fragment reads done
        commit_stage();
        __syncthreads();          // tiles visible
        if (kk + 32 < kEnd) load_stage(kk + 32);   // overlap HBM with WMMA below

        // A fragment (16x32 bf16, same K order as B):
        //  lanes 0-15 : row=lane,    K 0-15
        //  lanes 16-31: row=lane-16, K 16-31
        union { v4u u[2]; v16bf v; } af;
        {
            const unsigned* ap = ldsA + (mt * 16 + (lane & 15)) * 20 + ((lane >> 4) * 8);
            af.u[0] = *(const v4u*)(ap);
            af.u[1] = *(const v4u*)(ap + 4);
        }
        // B fragments: lanes 0-15: col N=lane, K 0-15; lanes 16-31: K 16-31
        #pragma unroll
        for (int t = 0; t < 4; ++t) {
            union { v4u u[2]; v16bf v; } bfm;
            const unsigned* bp = ldsB + ((ng + t) * 16 + (lane & 15)) * 20 + ((lane >> 4) * 8);
            bfm.u[0] = *(const v4u*)(bp);
            bfm.u[1] = *(const v4u*)(bp + 4);
            acc[t] = __builtin_amdgcn_wmma_f32_16x16x32_bf16(
                         false, af.v, false, bfm.v, (short)0, acc[t], false, false);
        }
    }

    // Store C (E x 64 x N), ISA 16x16 f32 D layout:
    //  vgpr j, lanes 0-15: M=j, N=lane; lanes 16-31: M=8+j, N=lane-16
    float* Ce = C + (long)blockIdx.z * partStride + (long)e * 64 * N;
    const int rowBase = mt * 16 + (lane >> 4) * 8;
    const int col0    = nBase + ng * 16 + (lane & 15);
    float* p = Ce + (long)rowBase * N + col0;
    #pragma unroll
    for (int t = 0; t < 4; ++t) {
        #pragma unroll
        for (int j = 0; j < 8; ++j) {
            float v = acc[t][j];
            if (RELU) v = v > 0.f ? v : 0.f;
            p[(long)j * N + t * 16] = v;
        }
    }
}

// Deterministic split-K combine: out[i] = sum_s part[s*stride + i]
template<int S>
__global__ __launch_bounds__(256) void combine_splits(
    const float* __restrict__ part, float* __restrict__ out,
    int total, long stride)
{
    int i = blockIdx.x * 256 + threadIdx.x;
    if (i < total) {
        float s = 0.f;
        #pragma unroll
        for (int j = 0; j < S; ++j) s += part[(long)j * stride + i];
        out[i] = s;
    }
}

// out[b,o] = sum_e scores[e] * h4[e,b,o]
__global__ __launch_bounds__(256) void reduce_experts(
    const float* __restrict__ h4, const float* __restrict__ scores,
    float* __restrict__ out, int total)
{
    int i = blockIdx.x * 256 + threadIdx.x;
    if (i < total) {
        float s = 0.f;
        #pragma unroll
        for (int e = 0; e < E_DIM; ++e)
            s += scores[e] * h4[(long)e * total + i];
        out[i] = s;
    }
}

extern "C" void kernel_launch(void* const* d_in, const int* in_sizes, int n_in,
                              void* d_out, int out_size, void* d_ws, size_t ws_size,
                              hipStream_t stream) {
    const float* x      = (const float*)d_in[0];
    const float* scores = (const float*)d_in[1];
    const float* delta  = (const float*)d_in[2];
    const float* W1     = (const float*)d_in[3];
    const float* W2     = (const float*)d_in[4];
    const float* W3     = (const float*)d_in[5];
    const float* W4     = (const float*)d_in[6];

    const long nSmall = (long)E_DIM * B_DIM * R_DIM;    // 131072
    const long nBig   = (long)E_DIM * B_DIM * H_DIM;    // 1048576
    float* h1 = (float*)d_ws;
    float* h2 = h1 + nSmall;
    float* h3 = h2 + nBig;
    float* h4 = h3 + nSmall;
    float* P  = h4 + nBig;                              // 8 split-K partial slices

    const long off1 = 0;
    const long off2 = (long)R_DIM * IN_DIM;
    const long off3 = off2 + (long)H_DIM * R_DIM;
    const long off4 = off3 + (long)R_DIM * H_DIM;

    dim3 blk(256);
    const int S = 8;  // K-splits for the skinny N=512 GEMMs (=> 128 WGs)

    // Layer 1: h1 = x * (W1+d1)^T   (K=4096, N=512, split-K)
    lora_gemm<0><<<dim3(R_DIM / 128, E_DIM, S), blk, 0, stream>>>(
        x, 0L, W1, delta, off1, P, nSmall, IN_DIM, R_DIM, IN_DIM / S);
    combine_splits<S><<<(int)((nSmall + 255) / 256), blk, 0, stream>>>(P, h1, (int)nSmall, nSmall);

    // Layer 2: h2 = relu(h1 * (W2+d2)^T)   (K=512, N=4096, direct + fused ReLU)
    lora_gemm<1><<<dim3(H_DIM / 128, E_DIM, 1), blk, 0, stream>>>(
        h1, (long)B_DIM * R_DIM, W2, delta, off2, h2, 0L, R_DIM, H_DIM, R_DIM);

    // Layer 3: h3 = h2 * (W3+d3)^T   (K=4096, N=512, split-K)
    lora_gemm<0><<<dim3(R_DIM / 128, E_DIM, S), blk, 0, stream>>>(
        h2, (long)B_DIM * H_DIM, W3, delta, off3, P, nSmall, H_DIM, R_DIM, H_DIM / S);
    combine_splits<S><<<(int)((nSmall + 255) / 256), blk, 0, stream>>>(P, h3, (int)nSmall, nSmall);

    // Layer 4: h4 = h3 * (W4+d4)^T   (K=512, N=4096, direct)
    lora_gemm<0><<<dim3(OUT_DIM / 128, E_DIM, 1), blk, 0, stream>>>(
        h3, (long)B_DIM * R_DIM, W4, delta, off4, h4, 0L, R_DIM, OUT_DIM, R_DIM);

    // Final: out = sum_e scores[e] * h4[e]
    const int totalOut = B_DIM * OUT_DIM;
    reduce_experts<<<(totalOut + 255) / 256, blk, 0, stream>>>(h4, scores, (float*)d_out, totalOut);
}
